// VarianceAdaptor_11106785427610
// MI455X (gfx1250) — matrix operand order repaired and checked by reading, not compiled, not run on hardware
//
#include <hip/hip_runtime.h>

// ---------------------------------------------------------------------------
// Problem constants (from the reference): B=32, L=512, H=F=256, K=3, NB=256,
// M=2048.  Conv work = 38.7 GFLOP (tiny for WMMA); mel output = 67MB f32 is
// the bandwidth floor (~11us @ 23.3TB/s).  Strategy: bf16 activations +
// v_wmma_f32_16x16x32_bf16 with fused bias+ReLU+LayerNorm epilogue so each
// conv layer touches memory exactly once in, once out.
// ---------------------------------------------------------------------------

#define Bn 32
#define Ln 512
#define Hn 256
#define Fn 256
#define Mn 2048
#define NBm1 255   // NB-1 bins

typedef __attribute__((ext_vector_type(16))) __bf16 bf16x16;
typedef __attribute__((ext_vector_type(8)))  __bf16 bf16x8;
typedef __attribute__((ext_vector_type(8)))  float  f32x8;

__device__ __forceinline__ unsigned short f2bf(float f) {
  unsigned int u = __float_as_uint(f);
  unsigned int r = u + 0x7FFFu + ((u >> 16) & 1u);   // round-to-nearest-even
  return (unsigned short)(r >> 16);
}
__device__ __forceinline__ float bf2f(unsigned short b) {
  return __uint_as_float(((unsigned int)b) << 16);
}

// ---------------------------------------------------------------------------
// Permute conv weight [3][256][256] f32 into the WMMA B-matrix (32x16 bf16)
// lane layout: wlay[t][kc][nf][lane][e], where K = kc*32 + (e&7) + 8*(lane>>4) + 16*(e>>3)
// and N = nf*16 + (lane&15).  After this, each lane's B operand for one WMMA
// is a single contiguous, 32B-aligned load.
// ---------------------------------------------------------------------------
__global__ void prep_w(const float* __restrict__ w, unsigned short* __restrict__ wlay) {
  int id = blockIdx.x * 256 + threadIdx.x;
  if (id >= 3 * 8 * 16 * 32 * 16) return;           // 196608
  int e    = id & 15;
  int lane = (id >> 4) & 31;
  int nf   = (id >> 9) & 15;
  int kc   = (id >> 13) & 7;
  int t    = id >> 16;
  int k = kc * 32 + (e & 7) + 8 * (lane >> 4) + 16 * (e >> 3);
  int n = nf * 16 + (lane & 15);
  wlay[id] = f2bf(w[(t * Fn + k) * Fn + n]);
}

__global__ void cast_bf(const float* __restrict__ in, unsigned short* __restrict__ out, int n) {
  int i = blockIdx.x * 256 + threadIdx.x;
  if (i < n) out[i] = f2bf(in[i]);
}

// ---------------------------------------------------------------------------
// Fused Conv1D(k=3, same) + bias + ReLU + LayerNorm.  One wave per 16 rows of
// (b,l); computes the full 16x256 output tile with 3*8*16 = 384 bf16 WMMAs.
// A-tile per-lane loads follow the ISA 16-bit A 16x32 layout:
//   lane m=lane&15, half=lane>>4, elements 0..7 -> K=8h..8h+7,
//   elements 8..15 -> K=8h+16..8h+23  (two contiguous 16B global loads).
// ---------------------------------------------------------------------------
__global__ __launch_bounds__(32)
void conv_relu_ln(const unsigned short* __restrict__ in,    // [B*L][256] bf16
                  const unsigned short* __restrict__ wlay,  // permuted weights
                  const float* __restrict__ bias,
                  const float* __restrict__ gamma,
                  const float* __restrict__ beta,
                  unsigned short* __restrict__ out)         // [B*L][256] bf16
{
  __shared__ float lds[16 * 256];
  const int lane = threadIdx.x;
  const int m  = lane & 15;
  const int hh = lane >> 4;
  const int r0 = blockIdx.x * 16;      // first (b,l) row of the tile
  const int b  = r0 / Ln;
  const int l0 = r0 - b * Ln;

  f32x8 acc[16];
#pragma unroll
  for (int nf = 0; nf < 16; ++nf) acc[nf] = {};

  for (int t = 0; t < 3; ++t) {
    const int lrow = l0 + m + t - 1;                 // 'SAME' pad=1
    const bool valid = (lrow >= 0) && (lrow < Ln);
    for (int kc = 0; kc < 8; ++kc) {
      union { bf16x16 v; bf16x8 h[2]; unsigned int u[8]; } A;
      if (valid) {
        const unsigned short* p = in + ((long)(b * Ln + lrow)) * Fn + kc * 32 + 8 * hh;
        A.h[0] = *(const bf16x8*)(p);                // K = 8h .. 8h+7
        A.h[1] = *(const bf16x8*)(p + 16);           // K = 8h+16 .. 8h+23
      } else {
#pragma unroll
        for (int q = 0; q < 8; ++q) A.u[q] = 0u;     // zero padding rows
      }
      const unsigned short* wp = wlay + (((t * 8 + kc) * 16) * 32 + lane) * 16;
#pragma unroll
      for (int nf = 0; nf < 16; ++nf) {
        bf16x16 Bv = *(const bf16x16*)(wp + nf * (32 * 16));
        acc[nf] = __builtin_amdgcn_wmma_f32_16x16x32_bf16(
            false, A.v, false, Bv, (short)0, acc[nf], false, false);
      }
    }
  }

  // D layout: lane n=lane&15, VGPR r -> row r + 8*(lane>>4), col nf*16+n.
  // Epilogue part 1: bias + ReLU into LDS (f32).
#pragma unroll
  for (int nf = 0; nf < 16; ++nf) {
    const int n = nf * 16 + m;
#pragma unroll
    for (int r = 0; r < 8; ++r) {
      const int row = r + 8 * hh;
      float v = acc[nf][r] + bias[n];
      lds[row * 256 + n] = v > 0.f ? v : 0.f;
    }
  }
  __syncthreads();

  // Epilogue part 2: LayerNorm per row; 2 lanes per row, shfl_xor combine.
  const int row = lane >> 1;
  const int c0  = (lane & 1) * 128;
  float s = 0.f, q = 0.f;
  for (int c = 0; c < 128; ++c) {
    const float v = lds[row * 256 + c0 + c];
    s += v; q += v * v;
  }
  s += __shfl_xor(s, 1, 32);
  q += __shfl_xor(q, 1, 32);
  const float mean = s * (1.0f / 256.0f);
  const float var  = q * (1.0f / 256.0f) - mean * mean;
  const float inv  = rsqrtf(var + 1e-5f);

  unsigned int* orow = (unsigned int*)(out + ((long)(r0 + row)) * Fn + c0);
  for (int c = 0; c < 128; c += 2) {
    const float v0 = (lds[row * 256 + c0 + c]     - mean) * inv * gamma[c0 + c]     + beta[c0 + c];
    const float v1 = (lds[row * 256 + c0 + c + 1] - mean) * inv * gamma[c0 + c + 1] + beta[c0 + c + 1];
    orow[c >> 1] = (unsigned int)f2bf(v0) | ((unsigned int)f2bf(v1) << 16);
  }
}

// Final Linear(256->1) + mask.  4.2M MACs total: pure bandwidth, VALU is fine.
__global__ void linear_mask(const unsigned short* __restrict__ h,
                            const float* __restrict__ lw,
                            const float* __restrict__ lb,
                            const unsigned char* __restrict__ mask,
                            float* __restrict__ out)
{
  const int i = blockIdx.x * 256 + threadIdx.x;     // (b,l) flat, 16384
  float acc = lb[0];
  const unsigned short* hp = h + (long)i * Fn;
  for (int f = 0; f < Fn; ++f) acc += bf2f(hp[f]) * lw[f];
  out[i] = mask[i] ? 0.f : acc;
}

// x += emb[searchsorted(bins, target)]  (side='left': count of bins < t)
__global__ void add_emb(const float* __restrict__ x,
                        const float* __restrict__ target,
                        const float* __restrict__ bins,   // 255 sorted values
                        const float* __restrict__ emb,    // [256][256]
                        float* __restrict__ xout,
                        unsigned short* __restrict__ xout_bf)
{
  __shared__ float sb[NBm1];
  __shared__ int sidx;
  const int i = blockIdx.x;          // (b,l)
  const int c = threadIdx.x;         // channel
  if (c < NBm1) sb[c] = bins[c];
  __syncthreads();
  if (c == 0) {
    const float t = target[i];
    int cnt = 0;
    for (int j = 0; j < NBm1; ++j) cnt += (sb[j] < t) ? 1 : 0;
    sidx = cnt;
  }
  __syncthreads();
  const float v = x[(long)i * Hn + c] + emb[(long)sidx * Hn + c];
  xout[(long)i * Hn + c] = v;
  if (xout_bf) xout_bf[(long)i * Hn + c] = f2bf(v);
}

// Per-batch cumsum of durations + mel_len = min(total, max_len)
__global__ void cumsum_dur(const int* __restrict__ dur, const int* __restrict__ maxlen,
                           int* __restrict__ cs, float* __restrict__ mel_len)
{
  const int b = blockIdx.x;
  if (threadIdx.x == 0) {
    int run = 0;
    for (int l = 0; l < Ln; ++l) { run += dur[b * Ln + l]; cs[b * Ln + l] = run; }
    const int ml = maxlen[0];
    mel_len[b] = (float)(run < ml ? run : ml);
  }
}

// Length regulator: frame j copies phoneme idx = searchsorted(cs, j, 'right');
// frames >= total are zero.  One block per (b, j) frame; 67MB write dominates.
__global__ void length_reg(const float* __restrict__ x2, const int* __restrict__ cs,
                           float* __restrict__ mel)
{
  __shared__ int sidx, stot;
  const int bj = blockIdx.x;
  const int b  = bj >> 11;           // / 2048
  const int j  = bj & 2047;
  if (threadIdx.x == 0) {
    const int* c = cs + b * Ln;
    const int tot = c[Ln - 1];
    int lo = 0, hi = Ln;               // first i with c[i] > j
    while (lo < hi) { const int mid = (lo + hi) >> 1; if (c[mid] <= j) lo = mid + 1; else hi = mid; }
    if (lo > Ln - 1) lo = Ln - 1;
    sidx = lo; stot = tot;
  }
  __syncthreads();
  const int c = threadIdx.x;
  const float v = (j < stot) ? x2[((long)(b * Ln + sidx)) * Hn + c] : 0.f;
  mel[((long)bj) * Hn + c] = v;
}

// ---------------------------------------------------------------------------
extern "C" void kernel_launch(void* const* d_in, const int* in_sizes, int n_in,
                              void* d_out, int out_size, void* d_ws, size_t ws_size,
                              hipStream_t stream)
{
  (void)in_sizes; (void)n_in; (void)out_size; (void)ws_size;

  // Input order = setup_inputs() dict order; nested param dicts flatten in
  // insertion order: w1,b1,g1,be1,w2,b2,g2,be2,lw,lb.
  const float*         x        = (const float*)d_in[0];
  const unsigned char* mask     = (const unsigned char*)d_in[1];   // bool [B,L]
  const int*           dur      = (const int*)d_in[2];
  const float*         pitch_t  = (const float*)d_in[3];
  const float*         energy_t = (const float*)d_in[4];
  const int*           maxlen   = (const int*)d_in[5];
  const int DUR = 6, PIT = 16, ENE = 26;
  auto P = [&](int base, int off) { return (const float*)d_in[base + off]; };
  const float* pitch_bins  = (const float*)d_in[36];
  const float* energy_bins = (const float*)d_in[37];
  const float* pitch_emb   = (const float*)d_in[38];
  const float* energy_emb  = (const float*)d_in[39];

  // Workspace carve-up (~70MB).
  char* ws = (char*)d_ws;
  size_t off = 0;
  auto alloc = [&](size_t bytes) -> void* {
    off = (off + 255) & ~(size_t)255;
    void* p = ws + off;
    off += bytes;
    return p;
  };
  unsigned short* wlay[6];
  for (int i = 0; i < 6; ++i) wlay[i] = (unsigned short*)alloc(196608 * 2);
  unsigned short* xbf  = (unsigned short*)alloc((size_t)Bn * Ln * Hn * 2);
  unsigned short* h1   = (unsigned short*)alloc((size_t)Bn * Ln * Fn * 2);
  unsigned short* h2   = (unsigned short*)alloc((size_t)Bn * Ln * Fn * 2);
  unsigned short* xpbf = (unsigned short*)alloc((size_t)Bn * Ln * Hn * 2);
  float* xp = (float*)alloc((size_t)Bn * Ln * Hn * 4);
  float* x2 = (float*)alloc((size_t)Bn * Ln * Hn * 4);
  int*   cs = (int*)alloc((size_t)Bn * Ln * 4);

  // Output carve-up: (mel, pitch_pred, energy_pred, log_dur, mel_len) flat.
  float* mel         = (float*)d_out;
  float* pitch_pred  = mel + (size_t)Bn * Mn * Hn;
  float* energy_pred = pitch_pred + Bn * Ln;
  float* log_dur     = energy_pred + Bn * Ln;
  float* mel_len     = log_dur + Bn * Ln;

  // Permute all six conv weights into WMMA B-lane layout (runs once per call).
  const float* wsrc[6] = { P(DUR,0), P(DUR,4), P(PIT,0), P(PIT,4), P(ENE,0), P(ENE,4) };
  for (int i = 0; i < 6; ++i)
    prep_w<<<768, 256, 0, stream>>>(wsrc[i], wlay[i]);

  cast_bf<<<(Bn * Ln * Hn + 255) / 256, 256, 0, stream>>>(x, xbf, Bn * Ln * Hn);

  const int gconv = Bn * Ln / 16;   // 1024 waves

  // Duration predictor
  conv_relu_ln<<<gconv, 32, 0, stream>>>(xbf, wlay[0], P(DUR,1), P(DUR,2), P(DUR,3), h1);
  conv_relu_ln<<<gconv, 32, 0, stream>>>(h1,  wlay[1], P(DUR,5), P(DUR,6), P(DUR,7), h2);
  linear_mask<<<Bn * Ln / 256, 256, 0, stream>>>(h2, P(DUR,8), P(DUR,9), mask, log_dur);

  // Pitch predictor (on original x)
  conv_relu_ln<<<gconv, 32, 0, stream>>>(xbf, wlay[2], P(PIT,1), P(PIT,2), P(PIT,3), h1);
  conv_relu_ln<<<gconv, 32, 0, stream>>>(h1,  wlay[3], P(PIT,5), P(PIT,6), P(PIT,7), h2);
  linear_mask<<<Bn * Ln / 256, 256, 0, stream>>>(h2, P(PIT,8), P(PIT,9), mask, pitch_pred);

  // x += pitch_emb[searchsorted(pitch_bins, pitch_target)]
  add_emb<<<Bn * Ln, 256, 0, stream>>>(x, pitch_t, pitch_bins, pitch_emb, xp, xpbf);

  // Energy predictor (on x + pitch embedding)
  conv_relu_ln<<<gconv, 32, 0, stream>>>(xpbf, wlay[4], P(ENE,1), P(ENE,2), P(ENE,3), h1);
  conv_relu_ln<<<gconv, 32, 0, stream>>>(h1,   wlay[5], P(ENE,5), P(ENE,6), P(ENE,7), h2);
  linear_mask<<<Bn * Ln / 256, 256, 0, stream>>>(h2, P(ENE,8), P(ENE,9), mask, energy_pred);

  // x2 = xp + energy_emb[searchsorted(energy_bins, energy_target)]
  add_emb<<<Bn * Ln, 256, 0, stream>>>(xp, energy_t, energy_bins, energy_emb, x2,
                                       (unsigned short*)nullptr);

  // Length regulate
  cumsum_dur<<<Bn, 32, 0, stream>>>(dur, maxlen, cs, mel_len);
  length_reg<<<Bn * Mn, 256, 0, stream>>>(x2, cs, mel);
}
